// GraphAttention_27479200760084
// MI455X (gfx1250) — matrix operand order repaired and checked by reading, not compiled, not run on hardware
//
#include <hip/hip_runtime.h>
#include <cstdint>
#include <cstddef>

typedef __attribute__((ext_vector_type(16))) _Float16 v16h;
typedef __attribute__((ext_vector_type(8)))  float    v8f;

namespace {

constexpr int kE   = 256;
constexpr int kS   = 2048;
constexpr int kB   = 4;
constexpr int kH   = 8;
constexpr int kD   = 32;
constexpr int kTok = kB * kS;                         // 8192
constexpr float kQScale = 0.17677669529663689f;       // 1/sqrt(32)

union V16H { v16h v; _Float16 h[16]; uint32_t u[8]; };

__device__ inline v8f wmma_f16(v16h a, v16h b, v8f c) {
  // v_wmma_f32_16x16x32_f16  D = A*B + C
  return __builtin_amdgcn_wmma_f32_16x16x32_f16(false, a, false, b, (short)0, c,
                                                false, false);
}

// ---- A operand (16xK, K=32), rows of length `ld` starting at p. f32 source.
// ISA layout: M = lane%16; lanes<16 hold K = kb+{0..7} and kb+{16..23},
// lanes>=16 hold K = kb+{8..15} and kb+{24..31}.
__device__ inline v16h load_a_f32(const float* __restrict__ p, int ld,
                                  int mbase, int kbase, int lane) {
  const int m  = mbase + (lane & 15);
  const int kb = kbase + ((lane & 16) ? 8 : 0);
  const float* row = p + (size_t)m * ld;
  V16H a;
#pragma unroll
  for (int t = 0; t < 4; ++t) {
    float2 lo = *(const float2*)(row + kb + 2 * t);
    float2 hi = *(const float2*)(row + kb + 16 + 2 * t);
    a.h[2 * t]     = (_Float16)lo.x;  a.h[2 * t + 1]     = (_Float16)lo.y;
    a.h[8 + 2 * t] = (_Float16)hi.x;  a.h[8 + 2 * t + 1] = (_Float16)hi.y;
  }
  return a.v;
}

// Same A layout, f16 source (2 halves per 4-byte load; coalesces to b128).
__device__ inline v16h load_a_f16(const _Float16* __restrict__ p, int ld,
                                  int mbase, int kbase, int lane) {
  const int m  = mbase + (lane & 15);
  const int kb = kbase + ((lane & 16) ? 8 : 0);
  const _Float16* row = p + (size_t)m * ld;
  V16H a;
#pragma unroll
  for (int t = 0; t < 4; ++t) {
    a.u[t]     = *(const uint32_t*)(row + kb + 2 * t);
    a.u[4 + t] = *(const uint32_t*)(row + kb + 16 + 2 * t);
  }
  return a.v;
}

// ---- B operand (K=32 x N=16): N = lane%16, lanes<16 hold K=kb+{0..7} and kb+{16..23},
// lanes>=16 hold K=kb+{8..15} and kb+{24..31}. Element B[k][n] taken from p[n*ld + k]
// (row-major source transposed on the fly).  f32 source.
__device__ inline v16h load_bt_f32(const float* __restrict__ p, int ld,
                                   int kbase, int nbase, int lane) {
  const int n  = nbase + (lane & 15);
  const int kb = kbase + ((lane & 16) ? 8 : 0);
  const float* row = p + (size_t)n * ld + kb;
  V16H b;
#pragma unroll
  for (int t = 0; t < 8; ++t) {
    float2 x = *(const float2*)(row + 2 * t + ((t & 4) ? 8 : 0));
    b.h[2 * t] = (_Float16)x.x;  b.h[2 * t + 1] = (_Float16)x.y;
  }
  return b.v;
}

// Same B layout, f16 source, consecutive K in memory (32B/lane -> b128).
__device__ inline v16h load_bt_f16(const _Float16* __restrict__ p, int ld,
                                   int kbase, int nbase, int lane) {
  const int n  = nbase + (lane & 15);
  const int kb = kbase + ((lane & 16) ? 8 : 0);
  const _Float16* row = p + (size_t)n * ld + kb;
  V16H b;
#pragma unroll
  for (int t = 0; t < 4; ++t) {
    b.u[t]     = *(const uint32_t*)(row + 2 * t);
    b.u[4 + t] = *(const uint32_t*)(row + 16 + 2 * t);
  }
  return b.v;
}

// A operand read back from a 16x32 f16 LDS tile (P transpose buffer).
__device__ inline v16h lds_load_a(const _Float16* buf, int lane) {
  const int m    = lane & 15;
  const int base = (lane & 16) ? 8 : 0;
  const _Float16* rowp = buf + m * 32;
  V16H a;
#pragma unroll
  for (int t = 0; t < 4; ++t) {
    a.u[t]     = *(const uint32_t*)(rowp + base + 2 * t);
    a.u[4 + t] = *(const uint32_t*)(rowp + base + 16 + 2 * t);
  }
  return a.v;
}

// ================= Kernel 1: fused QKV projection (f32 in -> f16 out) ======
// out[n,e] = (x[n,:] . W[e,:]) + b[e].  Q scaled by 1/sqrt(D) at store;
// V stored transposed as Vt[b][h][d][s] so P@V has consecutive-K loads.
__global__ void qkv_proj_kernel(const float* __restrict__ x,
                                const float* __restrict__ wq, const float* __restrict__ bq,
                                const float* __restrict__ wk, const float* __restrict__ bk,
                                const float* __restrict__ wv, const float* __restrict__ bv,
                                _Float16* __restrict__ q16,
                                _Float16* __restrict__ k16,
                                _Float16* __restrict__ vt16) {
  const int lane   = threadIdx.x;
  const int tile_m = (blockIdx.x >> 4) << 4;   // token tile (16 rows)
  const int tile_n = (blockIdx.x & 15) << 4;   // feature tile (16 cols)
  const int which  = blockIdx.y;               // 0=Q 1=K 2=V

  const float* w    = (which == 0) ? wq : (which == 1) ? wk : wv;
  const float* bias = (which == 0) ? bq : (which == 1) ? bk : bv;

  v8f acc = {};
#pragma unroll
  for (int kk = 0; kk < kE; kk += 32) {
    v16h a = load_a_f32(x, kE, tile_m, kk, lane);
    v16h b = load_bt_f32(w, kE, kk, tile_n, lane);
    acc = wmma_f16(a, b, acc);
  }

  const int n     = tile_n + (lane & 15);      // output feature e
  const float bn  = bias[n];
  const int m_loc = (lane & 16) ? 8 : 0;

  if (which == 2) {
    // Vt[((b*H + h)*D + d)*S + s]
    const int hh = n >> 5;
    const int dd = n & 31;
#pragma unroll
    for (int r = 0; r < 8; ++r) {
      const int m  = tile_m + r + m_loc;       // token index
      const int bi = m >> 11;                  // m / kS
      const int s  = m & (kS - 1);
      vt16[(((size_t)bi * kH + hh) * kD + dd) * kS + s] = (_Float16)(acc[r] + bn);
    }
  } else {
    _Float16* out     = (which == 0) ? q16 : k16;
    const float scale = (which == 0) ? kQScale : 1.0f;
#pragma unroll
    for (int r = 0; r < 8; ++r) {
      const int m = tile_m + r + m_loc;
      out[(size_t)m * kE + n] = (_Float16)((acc[r] + bn) * scale);
    }
  }
}

// ================= Kernel 2: masked-LeakyReLU attention ====================
// One wave: (b, h, 32 q-rows).  Scores are O(few sigma) so exp() needs no
// max-subtraction: per-lane partial softmax denominators, reduced once at end.
__global__ void graph_attn_kernel(const _Float16* __restrict__ q16,
                                  const _Float16* __restrict__ k16,
                                  const _Float16* __restrict__ vt16,
                                  const float* __restrict__ adj,
                                  _Float16* __restrict__ o16) {
  __shared__ _Float16 ldsP[2][16 * 32];        // P transpose buffers (2 q-tiles)

  const int lane  = threadIdx.x;
  const int qt    = blockIdx.x & 63;           // kS/32 = 64 macro q-tiles
  const int h     = (blockIdx.x >> 6) & 7;
  const int b     = blockIdx.x >> 9;
  const int qbase = qt << 5;                   // 32 q-rows
  const int m_loc = (lane & 16) ? 8 : 0;
  const int ncol  = lane & 15;

  const _Float16* qp  = q16  + (size_t)b * kS * kE + h * kD;
  const _Float16* kp  = k16  + (size_t)b * kS * kE + h * kD;
  const _Float16* vtp = vt16 + ((size_t)(b * kH + h)) * kD * kS;

  const v16h aq0 = load_a_f16(qp, kE, qbase,      0, lane);
  const v16h aq1 = load_a_f16(qp, kE, qbase + 16, 0, lane);

  float lsum0[8], lsum1[8];
#pragma unroll
  for (int r = 0; r < 8; ++r) { lsum0[r] = 0.0f; lsum1[r] = 0.0f; }
  v8f o00 = {}, o01 = {}, o10 = {}, o11 = {};

  for (int kb = 0; kb < kS; kb += 32) {
    v16h bk0 = load_bt_f16(kp, kE, 0, kb,      lane);
    v16h bk1 = load_bt_f16(kp, kE, 0, kb + 16, lane);
    v8f z = {};
    v8f s00 = wmma_f16(aq0, bk0, z);
    v8f s01 = wmma_f16(aq0, bk1, z);
    v8f s10 = wmma_f16(aq1, bk0, z);
    v8f s11 = wmma_f16(aq1, bk1, z);

#pragma unroll
    for (int r = 0; r < 8; ++r) {
      const int row0 = qbase + r + m_loc;
      const int row1 = row0 + 16;
      const float* ar0 = adj + (size_t)row0 * kS + kb + ncol;
      const float* ar1 = adj + (size_t)row1 * kS + kb + ncol;
      float x00 = s00[r] * ar0[0];   x00 = (x00 >= 0.0f) ? x00 : 0.2f * x00;
      float x01 = s01[r] * ar0[16];  x01 = (x01 >= 0.0f) ? x01 : 0.2f * x01;
      float x10 = s10[r] * ar1[0];   x10 = (x10 >= 0.0f) ? x10 : 0.2f * x10;
      float x11 = s11[r] * ar1[16];  x11 = (x11 >= 0.0f) ? x11 : 0.2f * x11;
      const float e00 = __expf(x00);
      const float e01 = __expf(x01);
      const float e10 = __expf(x10);
      const float e11 = __expf(x11);
      lsum0[r] += e00 + e01;
      lsum1[r] += e10 + e11;
      const int mrow = r + m_loc;
      ldsP[0][mrow * 32 + ncol]      = (_Float16)e00;
      ldsP[0][mrow * 32 + 16 + ncol] = (_Float16)e01;
      ldsP[1][mrow * 32 + ncol]      = (_Float16)e10;
      ldsP[1][mrow * 32 + 16 + ncol] = (_Float16)e11;
    }
    __syncthreads();

    const v16h ap0 = lds_load_a(ldsP[0], lane);
    const v16h ap1 = lds_load_a(ldsP[1], lane);
    const v16h bv0 = load_bt_f16(vtp, kS, kb, 0,  lane);
    const v16h bv1 = load_bt_f16(vtp, kS, kb, 16, lane);
    o00 = wmma_f16(ap0, bv0, o00);
    o01 = wmma_f16(ap0, bv1, o01);
    o10 = wmma_f16(ap1, bv0, o10);
    o11 = wmma_f16(ap1, bv1, o11);
    __syncthreads();
  }

  _Float16* op = o16 + (size_t)b * kS * kE + h * kD;
#pragma unroll
  for (int r = 0; r < 8; ++r) {
    float l0 = lsum0[r];
    float l1 = lsum1[r];
#pragma unroll
    for (int off = 1; off < 16; off <<= 1) {
      l0 += __shfl_xor(l0, off, 32);
      l1 += __shfl_xor(l1, off, 32);
    }
    const float inv0 = 1.0f / l0;
    const float inv1 = 1.0f / l1;
    const int srow0  = qbase + r + m_loc;
    const int srow1  = srow0 + 16;
    op[(size_t)srow0 * kE + ncol]      = (_Float16)(o00[r] * inv0);
    op[(size_t)srow0 * kE + 16 + ncol] = (_Float16)(o01[r] * inv0);
    op[(size_t)srow1 * kE + ncol]      = (_Float16)(o10[r] * inv1);
    op[(size_t)srow1 * kE + 16 + ncol] = (_Float16)(o11[r] * inv1);
  }
}

// ================= Kernel 3: output projection (f16 in -> f32 out) =========
__global__ void out_proj_kernel(const _Float16* __restrict__ x,
                                const float* __restrict__ w,
                                const float* __restrict__ bias,
                                float* __restrict__ out) {
  const int lane   = threadIdx.x;
  const int tile_m = (blockIdx.x >> 4) << 4;
  const int tile_n = (blockIdx.x & 15) << 4;

  v8f acc = {};
#pragma unroll
  for (int kk = 0; kk < kE; kk += 32) {
    v16h a = load_a_f16(x, kE, tile_m, kk, lane);
    v16h b = load_bt_f32(w, kE, kk, tile_n, lane);
    acc = wmma_f16(a, b, acc);
  }

  const int n     = tile_n + (lane & 15);
  const float bn  = bias[n];
  const int m_loc = (lane & 16) ? 8 : 0;
#pragma unroll
  for (int r = 0; r < 8; ++r) {
    const int m = tile_m + r + m_loc;
    out[(size_t)m * kE + n] = acc[r] + bn;
  }
}

} // namespace

extern "C" void kernel_launch(void* const* d_in, const int* in_sizes, int n_in,
                              void* d_out, int out_size, void* d_ws, size_t ws_size,
                              hipStream_t stream) {
  (void)in_sizes; (void)n_in; (void)out_size; (void)ws_size;

  const float* hidden = (const float*)d_in[0];
  const float* adj    = (const float*)d_in[1];
  const float* wq     = (const float*)d_in[2];
  const float* bq     = (const float*)d_in[3];
  const float* wk     = (const float*)d_in[4];
  const float* bk     = (const float*)d_in[5];
  const float* wv     = (const float*)d_in[6];
  const float* bv     = (const float*)d_in[7];
  const float* wo     = (const float*)d_in[8];
  const float* bo     = (const float*)d_in[9];
  float* out          = (float*)d_out;

  // Workspace: Q,K f16 [B*S,E]; Vt f16 [B,H,D,S]; attn f16 [B*S,E] = 16 MB
  _Float16* q16  = (_Float16*)d_ws;
  _Float16* k16  = q16  + (size_t)kTok * kE;
  _Float16* vt16 = k16  + (size_t)kTok * kE;
  _Float16* a16  = vt16 + (size_t)kTok * kE;

  const int tiles = (kTok / 16) * (kE / 16);   // 8192

  dim3 gqkv(tiles, 3);
  qkv_proj_kernel<<<gqkv, 32, 0, stream>>>(hidden, wq, bq, wk, bk, wv, bv,
                                           q16, k16, vt16);

  graph_attn_kernel<<<kB * kH * (kS / 32), 32, 0, stream>>>(q16, k16, vt16, adj, a16);

  out_proj_kernel<<<tiles, 32, 0, stream>>>(a16, wo, bo, out);
}
